// Attention_47012712022149
// MI455X (gfx1250) — compile-verified, matrix operand order checked by reading, not run on hardware
//
#include <hip/hip_runtime.h>

// MI455X / gfx1250: wave32, WMMA 16x16x32 bf16 -> f32, async global->LDS copies.
typedef __bf16 v8bf  __attribute__((ext_vector_type(8)));
typedef __bf16 v16bf __attribute__((ext_vector_type(16)));
typedef float  v8f   __attribute__((ext_vector_type(8)));

#define WMMA_BF16(a, b, c)                                                    \
  __builtin_amdgcn_wmma_f32_16x16x32_bf16(false, (a), false, (b), (short)0,   \
                                          (c), false, false)

static __device__ __forceinline__ v16bf cat8(v8bf lo, v8bf hi) {
  return __builtin_shufflevector(lo, hi, 0, 1, 2, 3, 4, 5, 6, 7, 8, 9, 10, 11,
                                 12, 13, 14, 15);
}

// Async DMA: 16B per lane, global -> LDS, tracked by ASYNCcnt (no VGPR bounce).
static __device__ __forceinline__ void async_copy_b128(const __bf16* gsrc,
                                                       __bf16* ldst) {
  unsigned dst = (unsigned)(uintptr_t)ldst;  // low 32 bits = DS offset
  unsigned long long src = (unsigned long long)(uintptr_t)gsrc;
  asm volatile("global_load_async_to_lds_b128 %0, %1, off"
               :: "v"(dst), "v"(src) : "memory");
}
static __device__ __forceinline__ void wait_async0() {
  asm volatile("s_wait_asynccnt 0x0" ::: "memory");
}

// ---------------------------------------------------------------------------
// Stage 0: fp32 -> bf16 downconvert (vectorized, n divisible by 2048)
// ---------------------------------------------------------------------------
__global__ __launch_bounds__(256) void cvt_f32_bf16(
    const float* __restrict__ in, __bf16* __restrict__ out, int n) {
  int idx = (blockIdx.x * 256 + threadIdx.x) * 8;
  if (idx >= n) return;
  float4 a = *(const float4*)(in + idx);
  float4 b = *(const float4*)(in + idx + 4);
  v8bf o = {(__bf16)a.x, (__bf16)a.y, (__bf16)a.z, (__bf16)a.w,
            (__bf16)b.x, (__bf16)b.y, (__bf16)b.z, (__bf16)b.w};
  *(v8bf*)(out + idx) = o;
}

// ---------------------------------------------------------------------------
// Stage 1 & 3: bf16 GEMM, 128x128 block tile, 8 waves, K-step 32,
// double-buffered LDS, async A-tile copies pipelined against WMMA.
// mode 0: C -> f32 outf.   mode 1: QKV epilogue (Q scaled, V transposed).
// ---------------------------------------------------------------------------
__global__ __launch_bounds__(256) void gemm_bf16_wmma(
    const __bf16* __restrict__ A, const __bf16* __restrict__ Bw, int M, int N,
    int K, int mode, float* __restrict__ outf, __bf16* __restrict__ qg,
    __bf16* __restrict__ kg, __bf16* __restrict__ vtg) {
  constexpr int LDT = 48;                 // 96B row stride: aligned b128 chunks
  __shared__ __bf16 As[2][128 * LDT];
  __shared__ __bf16 Bs[2][128 * LDT];     // B^T tile: [n][k]

  const int tid = threadIdx.x, lane = tid & 31, wv = tid >> 5;
  const int bm = blockIdx.y * 128, bn = blockIdx.x * 128;
  const int wm = (wv >> 1) * 32, wn = (wv & 1) * 64;
  const int half = lane >> 4, r = lane & 15, koff = half * 8;

  v8f acc[2][4] = {};

  // Issue async copy of the A tile (128x32 bf16, 512 x 16B, 2 chunks/thread).
  auto issueA = [&](int k0, int buf) {
#pragma unroll
    for (int it = 0; it < 2; ++it) {
      int slot = it * 256 + tid;
      int ar = slot >> 2, ac = (slot & 3) * 8;
      async_copy_b128(A + (size_t)(bm + ar) * K + k0 + ac,
                      &As[buf][ar * LDT + ac]);
    }
  };
  // Global loads of the B tile (coalesced v8bf, kept in regs).
  auto loadB = [&](int k0, v8bf* bv) {
#pragma unroll
    for (int it = 0; it < 2; ++it) {
      int slot = it * 256 + tid;
      int br = slot >> 4, bc = (slot & 15) * 8;
      bv[it] = *(const v8bf*)(Bw + (size_t)(k0 + br) * N + bn + bc);
    }
  };
  // Transposed scatter of the B tile into LDS.
  auto storeB = [&](int buf, const v8bf* bv) {
#pragma unroll
    for (int it = 0; it < 2; ++it) {
      int slot = it * 256 + tid;
      int br = slot >> 4, bc = (slot & 15) * 8;
#pragma unroll
      for (int e = 0; e < 8; ++e) Bs[buf][(bc + e) * LDT + br] = bv[it][e];
    }
  };

  v8bf bv[2];
  issueA(0, 0);
  loadB(0, bv);
  storeB(0, bv);

  for (int k0 = 0; k0 < K; k0 += 32) {
    const int cur = (k0 >> 5) & 1;
    wait_async0();       // own async copies for `cur` have landed in LDS
    __syncthreads();     // emits s_wait_dscnt 0 first: B stores visible too
    const bool more = (k0 + 32) < K;
    if (more) {
      issueA(k0 + 32, cur ^ 1);  // DMA overlaps with the WMMA work below
      loadB(k0 + 32, bv);        // loadcnt drains during compute
    }

    v16bf af[2], bfr[4];
#pragma unroll
    for (int mi = 0; mi < 2; ++mi) {
      const __bf16* p = &As[cur][(wm + mi * 16 + r) * LDT + koff];
      af[mi] = cat8(*(const v8bf*)p, *(const v8bf*)(p + 16));
    }
#pragma unroll
    for (int ni = 0; ni < 4; ++ni) {
      const __bf16* p = &Bs[cur][(wn + ni * 16 + r) * LDT + koff];
      bfr[ni] = cat8(*(const v8bf*)p, *(const v8bf*)(p + 16));
    }
#pragma unroll
    for (int mi = 0; mi < 2; ++mi)
#pragma unroll
      for (int ni = 0; ni < 4; ++ni)
        acc[mi][ni] = WMMA_BF16(af[mi], bfr[ni], acc[mi][ni]);

    if (more) storeB(cur ^ 1, bv);
  }

  // Epilogue. C/D layout: lane<16 -> (M=j, N=lane); lane>=16 -> (M=j+8, N=lane-16)
#pragma unroll
  for (int mi = 0; mi < 2; ++mi)
#pragma unroll
    for (int ni = 0; ni < 4; ++ni)
#pragma unroll
      for (int j = 0; j < 8; ++j) {
        int gr = bm + wm + mi * 16 + j + half * 8;
        int gc = bn + wn + ni * 16 + r;
        float v = acc[mi][ni][j];
        if (mode == 0) {
          outf[(size_t)gr * N + gc] = v;
        } else {
          int b = gr >> 11, s = gr & 2047;
          int seg = gc >> 10, hc = gc & 1023;
          int h = hc >> 6, dd = hc & 63;
          size_t bh = (size_t)(b * 16 + h);
          if (seg == 0)
            qg[(bh * 2048 + s) * 64 + dd] = (__bf16)(v * 0.125f);  // dh^-0.5
          else if (seg == 1)
            kg[(bh * 2048 + s) * 64 + dd] = (__bf16)v;
          else
            vtg[(bh * 64 + dd) * 2048 + s] = (__bf16)v;  // V transposed
        }
      }
}

// ---------------------------------------------------------------------------
// Stage 2: causal flash attention. One wave = 16 Q rows; block = 128 Q rows
// sharing the same (b,h) K/V stream for cache reuse. KV chunk = 32 keys.
// ---------------------------------------------------------------------------
__global__ __launch_bounds__(256) void flash_attn_wmma(
    const __bf16* __restrict__ qg, const __bf16* __restrict__ kg,
    const __bf16* __restrict__ vtg, __bf16* __restrict__ aout) {
  constexpr int LP = 48;
  __shared__ __bf16 Ps[8 * 16 * LP];  // per-wave P re-layout buffer
  const int lane = threadIdx.x & 31, wv = threadIdx.x >> 5;
  const int half = lane >> 4, r = lane & 15, koff = half * 8;
  const int bh = blockIdx.y;  // b*16 + h
  const int q0 = blockIdx.x * 128 + wv * 16;

  const __bf16* Q = qg + ((size_t)bh * 2048 + q0) * 64;
  const __bf16* Kp = kg + (size_t)bh * 2048 * 64;
  const __bf16* Vt = vtg + (size_t)bh * 64 * 2048;
  __bf16* Pw = &Ps[wv * 16 * LP];

  // Q tile 16x64 -> two A-fragments, kept in registers for the whole loop
  v16bf qa[2];
#pragma unroll
  for (int kk = 0; kk < 2; ++kk) {
    const __bf16* p = Q + r * 64 + kk * 32 + koff;
    qa[kk] = cat8(*(const v8bf*)p, *(const v8bf*)(p + 16));
  }

  v8f o[4] = {};
  float mstat[8], lstat[8];
#pragma unroll
  for (int j = 0; j < 8; ++j) { mstat[j] = -3.0e38f; lstat[j] = 0.0f; }

  const int nsteps = (q0 + 16 + 31) >> 5;  // causal: keys 0..q0+15
  for (int st = 0; st < nsteps; ++st) {
    const int kv0 = st * 32;
    if (st + 1 < nsteps)  // prefetch next K chunk (32 rows x 128B)
      __builtin_prefetch(Kp + (size_t)(kv0 + 32 + lane) * 64, 0, 0);

    // S = Q @ K^T : 2 key sub-tiles x 2 inner-K steps
    v8f sc[2] = {};
#pragma unroll
    for (int t = 0; t < 2; ++t)
#pragma unroll
      for (int kk = 0; kk < 2; ++kk) {
        const __bf16* p =
            Kp + (size_t)(kv0 + t * 16 + r) * 64 + kk * 32 + koff;
        v16bf kf = cat8(*(const v8bf*)p, *(const v8bf*)(p + 16));
        sc[t] = WMMA_BF16(qa[kk], kf, sc[t]);
      }

    if (st == nsteps - 1) {  // only the diagonal chunk needs the causal mask
#pragma unroll
      for (int t = 0; t < 2; ++t)
#pragma unroll
        for (int j = 0; j < 8; ++j)
          if (kv0 + t * 16 + r > q0 + j + half * 8) sc[t][j] = -1.0e30f;
    }

    // Online softmax; each lane holds stats for rows (j + half*8)
#pragma unroll
    for (int j = 0; j < 8; ++j) {
      float mx = fmaxf(sc[0][j], sc[1][j]);
      mx = fmaxf(mx, __shfl_xor(mx, 1));
      mx = fmaxf(mx, __shfl_xor(mx, 2));
      mx = fmaxf(mx, __shfl_xor(mx, 4));
      mx = fmaxf(mx, __shfl_xor(mx, 8));
      float mn = fmaxf(mstat[j], mx);
      float alpha = __expf(mstat[j] - mn);
      mstat[j] = mn;
      float p0 = __expf(sc[0][j] - mn);
      float p1 = __expf(sc[1][j] - mn);
      sc[0][j] = p0;
      sc[1][j] = p1;
      float rs = p0 + p1;
      rs += __shfl_xor(rs, 1);
      rs += __shfl_xor(rs, 2);
      rs += __shfl_xor(rs, 4);
      rs += __shfl_xor(rs, 8);
      lstat[j] = lstat[j] * alpha + rs;
      o[0][j] *= alpha; o[1][j] *= alpha; o[2][j] *= alpha; o[3][j] *= alpha;
    }

    // Re-layout P (C-layout -> A-fragment) through per-wave LDS
#pragma unroll
    for (int t = 0; t < 2; ++t)
#pragma unroll
      for (int j = 0; j < 8; ++j)
        Pw[(j + half * 8) * LP + t * 16 + r] = (__bf16)sc[t][j];
    asm volatile("s_wait_dscnt 0x0" ::: "memory");

    const __bf16* pp = &Pw[r * LP + koff];
    v16bf pf = cat8(*(const v8bf*)pp, *(const v8bf*)(pp + 16));

    // O += P @ V  (V stored transposed -> contiguous B-fragment loads)
#pragma unroll
    for (int ni = 0; ni < 4; ++ni) {
      const __bf16* vp = Vt + (size_t)(ni * 16 + r) * 2048 + kv0 + koff;
      v16bf vf = cat8(*(const v8bf*)vp, *(const v8bf*)(vp + 16));
      o[ni] = WMMA_BF16(pf, vf, o[ni]);
    }
  }

  // Normalize and store to [B, S, H*dh] bf16
  const int b = bh >> 4, h = bh & 15;
#pragma unroll
  for (int j = 0; j < 8; ++j) {
    float inv = 1.0f / lstat[j];
    int row = q0 + j + half * 8;
    size_t base = ((size_t)b * 2048 + row) * 1024 + h * 64 + r;
#pragma unroll
    for (int ni = 0; ni < 4; ++ni)
      aout[base + ni * 16] = (__bf16)(o[ni][j] * inv);
  }
}

// ---------------------------------------------------------------------------
extern "C" void kernel_launch(void* const* d_in, const int* in_sizes, int n_in,
                              void* d_out, int out_size, void* d_ws,
                              size_t ws_size, hipStream_t stream) {
  const float* x = (const float*)d_in[0];     // [2, 2048, 1024]
  const float* wqkv = (const float*)d_in[1];  // [1024, 3072]
  const float* wout = (const float*)d_in[2];  // [1024, 1024]
  float* out = (float*)d_out;                 // [2, 2048, 1024]

  // Workspace layout (bf16), ~48 MB total
  __bf16* xb = (__bf16*)d_ws;                    // 4096*1024
  __bf16* wqkvb = xb + (size_t)4096 * 1024;      // 1024*3072
  __bf16* woutb = wqkvb + (size_t)1024 * 3072;   // 1024*1024
  __bf16* qg = woutb + (size_t)1024 * 1024;      // [32][2048][64]
  __bf16* kg = qg + (size_t)32 * 2048 * 64;      // [32][2048][64]
  __bf16* vtg = kg + (size_t)32 * 2048 * 64;     // [32][64][2048]
  __bf16* aout = vtg + (size_t)32 * 2048 * 64;   // 4096*1024

  cvt_f32_bf16<<<4194304 / 2048, 256, 0, stream>>>(x, xb, 4194304);
  cvt_f32_bf16<<<3145728 / 2048, 256, 0, stream>>>(wqkv, wqkvb, 3145728);
  cvt_f32_bf16<<<1048576 / 2048, 256, 0, stream>>>(wout, woutb, 1048576);

  // QKV projection: [4096,1024] @ [1024,3072] -> Q/K/V^T (bf16)
  gemm_bf16_wmma<<<dim3(24, 32), 256, 0, stream>>>(xb, wqkvb, 4096, 3072, 1024,
                                                   1, nullptr, qg, kg, vtg);
  // Causal flash attention, per (b,h) x 128-row Q tiles
  flash_attn_wmma<<<dim3(16, 32), 256, 0, stream>>>(qg, kg, vtg, aout);
  // Output projection: [4096,1024] @ [1024,1024] -> f32 out
  gemm_bf16_wmma<<<dim3(8, 32), 256, 0, stream>>>(aout, woutb, 4096, 1024,
                                                  1024, 0, out, nullptr,
                                                  nullptr, nullptr);
}
